// MLPMoE_62156766708054
// MI455X (gfx1250) — compile-verified, matrix operand order checked by reading, not run on hardware
//
#include <hip/hip_runtime.h>
#include <hip/hip_bf16.h>

typedef __attribute__((ext_vector_type(16))) _Float16 v16h;
typedef __attribute__((ext_vector_type(8)))  _Float16 v8h;
typedef __attribute__((ext_vector_type(8)))  float    v8f;

#define STR  136   // activation/weight LDS row stride (halfs), mult of 8 -> 16B rows
#define XSTR 40    // layer-0 (K=32) LDS row stride (halfs), 80B rows, bank-spread
#define BM   128   // points per block
#define HH   128   // hidden width
#define NE   8     // experts

// ---- smem layout (bytes) ----
#define OFF_HBUF   0                       // 128*136*2 = 34816
#define OFF_WBUF1  34816                   // W1 tile
#define OFF_WBUF2  69632                   // W2 tile
#define OFF_XBUF   104448                  // 128*40*2 = 10240
#define OFF_W0BUF  114688                  // 128*40*2 = 10240
#define OFF_FLOATS 124928                  // 2 x (bias0|bias1|bias2|wo) = 2*2048
#define SMEM_BYTES (OFF_FLOATS + 2 * 4 * 128 * 4)   // 129024

union FragU { v16h v; uint4 u[2]; };

// 16x32 f16 fragment in the ISA A-layout (also B = W^T: lane n holds row n of W
// along K). lanes 0-15 -> K {0..7,16..23}, lanes 16-31 -> K {8..15,24..31}.
template <int LSTR>
__device__ __forceinline__ v16h load_frag(const _Float16* buf, int row0, int kbase, int lane) {
  const int r    = lane & 15;
  const int half = (lane >> 4) & 1;
  const _Float16* p = buf + (row0 + r) * LSTR + kbase + half * 8;
  FragU f;
  f.u[0] = *(const uint4*)(p);
  f.u[1] = *(const uint4*)(p + 16);
  return f.v;
}

// ---- CDNA5 async global->LDS copy (ASYNCcnt-tracked, bypasses VGPR data path) ----
__device__ __forceinline__ void async_copy_b128(void* lds_dst, const void* gsrc) {
#if defined(__gfx1250__)
  unsigned int       loff = (unsigned int)(unsigned long long)(uintptr_t)lds_dst;
  unsigned long long g    = (unsigned long long)(uintptr_t)gsrc;
  asm volatile("global_load_async_to_lds_b128 %0, %1, off" :: "v"(loff), "v"(g) : "memory");
#endif
}
__device__ __forceinline__ void wait_async() {
#if defined(__gfx1250__)
  asm volatile("s_wait_asynccnt 0x0" ::: "memory");
#endif
}

// One layer for this wave's 16-row strip: hbuf(rows) := relu(A @ W^T + b).
// A rows are wave-private; B fragments are double-buffered across N-tiles.
template <int KSTEPS, int ASTR>
__device__ __forceinline__ void mlp_layer(const _Float16* abuf, _Float16* hbuf,
                                          const _Float16* wbuf, const float* bias,
                                          int wave, int lane) {
  const int row0  = wave * 16;
  const int nlane = lane & 15;
  const int mhalf = (lane >> 4) & 1;

  v16h a[KSTEPS];
#pragma unroll
  for (int k = 0; k < KSTEPS; ++k) a[k] = load_frag<ASTR>(abuf, row0, 32 * k, lane);

  const int BSTR = (KSTEPS == 1) ? XSTR : STR;
  v16h b[KSTEPS], bn[KSTEPS];
#pragma unroll
  for (int k = 0; k < KSTEPS; ++k) b[k] = load_frag<BSTR>(wbuf, 0, 32 * k, lane);

#pragma unroll
  for (int nt = 0; nt < 8; ++nt) {
    if (nt < 7) {                       // prefetch next tile's B fragments
#pragma unroll
      for (int k = 0; k < KSTEPS; ++k) bn[k] = load_frag<BSTR>(wbuf, (nt + 1) * 16, 32 * k, lane);
    }
    v8f c = {0.f, 0.f, 0.f, 0.f, 0.f, 0.f, 0.f, 0.f};
#pragma unroll
    for (int k = 0; k < KSTEPS; ++k)
      c = __builtin_amdgcn_wmma_f32_16x16x32_f16(false, a[k], false, b[k], (short)0, c, false, false);

    const float bnv = bias[nt * 16 + nlane];
    _Float16* dst = hbuf + nt * 16 + nlane;
#pragma unroll
    for (int v = 0; v < 8; ++v) {       // C/D layout: VGPR v -> M = v + 8*mhalf
      float x = c[v] + bnv;
      x = x > 0.f ? x : 0.f;
      dst[(row0 + v + 8 * mhalf) * STR] = (_Float16)x;
    }
#pragma unroll
    for (int k = 0; k < KSTEPS; ++k) b[k] = bn[k];
  }
}

// Pre-pass: one-time f32 -> f16 weight conversion into workspace.
// w0h: [E][128][32] zero-padded along K; w1h/w2h: [E][128][128].
__global__ void __launch_bounds__(256)
convert_weights(const float* __restrict__ W0, const float* __restrict__ W1,
                const float* __restrict__ W2, _Float16* __restrict__ w0h,
                _Float16* __restrict__ w1h, _Float16* __restrict__ w2h) {
  const int i = blockIdx.x * 256 + threadIdx.x;
  if (i < NE * HH * HH) {
    w1h[i] = (_Float16)W1[i];
    w2h[i] = (_Float16)W2[i];
  }
  if (i < NE * HH) {                    // one thread per W0 row
    const float* s = W0 + i * 3;
    _Float16*    d = w0h + i * 32;
    d[0] = (_Float16)s[0]; d[1] = (_Float16)s[1]; d[2] = (_Float16)s[2];
#pragma unroll
    for (int k = 3; k < 32; ++k) d[k] = (_Float16)0.f;
  }
}

template <bool ASYNCW>
__global__ void __launch_bounds__(256)
moe_mlp_kernel(const float* __restrict__ coords,
               const float* __restrict__ W0, const float* __restrict__ b0,
               const float* __restrict__ W1, const float* __restrict__ b1,
               const float* __restrict__ W2, const float* __restrict__ b2,
               const float* __restrict__ Wo, const float* __restrict__ bo,
               const _Float16* __restrict__ w0h, const _Float16* __restrict__ w1h,
               const _Float16* __restrict__ w2h,
               float* __restrict__ out, int N) {
  extern __shared__ unsigned char smem_raw[];
  _Float16* hbuf  = (_Float16*)(smem_raw + OFF_HBUF);
  _Float16* wbuf1 = (_Float16*)(smem_raw + OFF_WBUF1);
  _Float16* wbuf2 = (_Float16*)(smem_raw + OFF_WBUF2);
  _Float16* xbuf  = (_Float16*)(smem_raw + OFF_XBUF);
  _Float16* w0buf = (_Float16*)(smem_raw + OFF_W0BUF);
  float*    flbase = (float*)(smem_raw + OFF_FLOATS);

  const int t    = threadIdx.x;
  const int lane = t & 31;
  const int wave = t >> 5;
  const int base = blockIdx.x * BM;

  // One-time: coords tile -> zero-padded f16 A-tile for the K=32 layer-0 WMMA.
  if (t < BM) {
    const int g = base + t;
    float x = 0.f, y = 0.f, z = 0.f;
    if (g < N) { x = coords[g * 3]; y = coords[g * 3 + 1]; z = coords[g * 3 + 2]; }
    _Float16* xr = xbuf + t * XSTR;
    xr[0] = (_Float16)x; xr[1] = (_Float16)y; xr[2] = (_Float16)z;
#pragma unroll
    for (int k = 3; k < 32; ++k) xr[k] = (_Float16)0.f;
  }

  float occ = -3.402823466e38f;
  const int orow  = t >> 1;
  const int ohalf = t & 1;

  for (int e = 0; e < NE; ++e) {
    // parity-double-buffered small-vector block: bias0|bias1|bias2|wo
    float* fl     = flbase + (e & 1) * 512;
    float* bias0  = fl;
    float* bias1  = fl + 128;
    float* bias2  = fl + 256;
    float* wo_lds = fl + 384;

    // ---- stage EVERYTHING for this expert up-front (no mid-expert stalls) ----
    if (ASYNCW) {
      {  // W0: 128 rows x 64B, 4 chunks/row
        const int chunk = t & 3, rip = t >> 2;
#pragma unroll
        for (int p = 0; p < 2; ++p) {
          const int row = p * 64 + rip;
          async_copy_b128(w0buf + row * XSTR + chunk * 8,
                          w0h + ((size_t)(e * HH + row)) * 32 + chunk * 8);
        }
      }
      {  // W1 and W2: 128 rows x 256B each, 16 chunks/row
        const int chunk = t & 15, rip = t >> 4;
#pragma unroll
        for (int p = 0; p < 8; ++p) {
          const int row = p * 16 + rip;
          async_copy_b128(wbuf1 + row * STR + chunk * 8,
                          w1h + ((size_t)(e * HH + row)) * HH + chunk * 8);
          async_copy_b128(wbuf2 + row * STR + chunk * 8,
                          w2h + ((size_t)(e * HH + row)) * HH + chunk * 8);
        }
      }
      if (t < 128) {  // biases + Wo: one b128 per thread into contiguous block
        const int grp = t >> 5, ch = t & 31;
        const float* g = (grp == 0) ? (b0 + e * HH)
                       : (grp == 1) ? (b1 + e * HH)
                       : (grp == 2) ? (b2 + e * HH)
                                    : (Wo + e * HH);
        async_copy_b128(fl + grp * 128 + ch * 4, g + ch * 4);
      }
      wait_async();
    } else {
      if (t < HH) {  // W0 cvt fallback (zero-padded row)
        const float* s = W0 + (e * HH + t) * 3;
        _Float16*    d = w0buf + t * XSTR;
        d[0] = (_Float16)s[0]; d[1] = (_Float16)s[1]; d[2] = (_Float16)s[2];
#pragma unroll
        for (int k = 3; k < 32; ++k) d[k] = (_Float16)0.f;
      }
      const float* ws1 = W1 + (size_t)e * HH * HH;
      const float* ws2 = W2 + (size_t)e * HH * HH;
      const int col = t & 127, r0 = t >> 7;
#pragma unroll 4
      for (int i = 0; i < 64; ++i) {
        const int r = r0 + 2 * i;
        wbuf1[r * STR + col] = (_Float16)ws1[r * HH + col];
        wbuf2[r * STR + col] = (_Float16)ws2[r * HH + col];
      }
      if (t < HH) {
        bias0[t]  = b0[e * HH + t];
        bias1[t]  = b1[e * HH + t];
        bias2[t]  = b2[e * HH + t];
        wo_lds[t] = Wo[e * HH + t];
      }
    }
    __syncthreads();   // staging visible; prev expert's output reads of hbuf done

    // ---- three layers back-to-back: activation rows are wave-private ----
    mlp_layer<1, XSTR>(xbuf, hbuf, w0buf, bias0, wave, lane);   // h0
    mlp_layer<4, STR >(hbuf, hbuf, wbuf1, bias1, wave, lane);   // h1
    mlp_layer<4, STR >(hbuf, hbuf, wbuf2, bias2, wave, lane);   // h2

    __syncthreads();   // h2 visible across waves for the output stage

    // ---- output: 2 threads/row, vectorized 64-wide dots, shuffle-combine, max ----
    {
      const _Float16* hr = hbuf + orow * STR + ohalf * 64;
      const float*    wr = wo_lds + ohalf * 64;
      float s = 0.f;
#pragma unroll
      for (int c8 = 0; c8 < 8; ++c8) {
        v8h hv = *(const v8h*)(hr + c8 * 8);
#pragma unroll
        for (int j = 0; j < 8; ++j) s += (float)hv[j] * wr[c8 * 8 + j];
      }
      s += __shfl_xor(s, 1, 32);        // wave32 pair combine
      const float o = s + bo[e];
      occ = occ > o ? occ : o;
    }
    // next expert's stage barrier protects hbuf; float block is parity-buffered
  }
  if (ohalf == 0 && base + orow < N) out[base + orow] = occ;
}

extern "C" void kernel_launch(void* const* d_in, const int* in_sizes, int n_in,
                              void* d_out, int out_size, void* d_ws, size_t ws_size,
                              hipStream_t stream) {
  const float* coords = (const float*)d_in[0];
  const float* W0 = (const float*)d_in[1];
  const float* b0 = (const float*)d_in[2];
  const float* W1 = (const float*)d_in[3];
  const float* b1 = (const float*)d_in[4];
  const float* W2 = (const float*)d_in[5];
  const float* b2 = (const float*)d_in[6];
  const float* Wo = (const float*)d_in[7];
  const float* bo = (const float*)d_in[8];
  float* out = (float*)d_out;

  const int N = in_sizes[0] / 3;
  const int blocks = (N + BM - 1) / BM;

  // Workspace layout: w0h [E][128][32] | w1h [E][128][128] | w2h [E][128][128]
  const size_t n_w0h = (size_t)NE * HH * 32;
  const size_t n_w1h = (size_t)NE * HH * HH;
  const size_t need  = (n_w0h + 2 * n_w1h) * sizeof(_Float16);   // ~576 KB

  (void)hipFuncSetAttribute((const void*)moe_mlp_kernel<true>,
                            hipFuncAttributeMaxDynamicSharedMemorySize, SMEM_BYTES);
  (void)hipFuncSetAttribute((const void*)moe_mlp_kernel<false>,
                            hipFuncAttributeMaxDynamicSharedMemorySize, SMEM_BYTES);

  if (ws_size >= need) {
    _Float16* w0h = (_Float16*)d_ws;
    _Float16* w1h = w0h + n_w0h;
    _Float16* w2h = w1h + n_w1h;
    convert_weights<<<(NE * HH * HH + 255) / 256, 256, 0, stream>>>(W0, W1, W2, w0h, w1h, w2h);
    moe_mlp_kernel<true><<<blocks, 256, SMEM_BYTES, stream>>>(
        coords, W0, b0, W1, b1, W2, b2, Wo, bo, w0h, w1h, w2h, out, N);
  } else {
    moe_mlp_kernel<false><<<blocks, 256, SMEM_BYTES, stream>>>(
        coords, W0, b0, W1, b1, W2, b2, Wo, bo, nullptr, nullptr, nullptr, out, N);
  }
}
